// EmbeddingSpace_13400297963894
// MI455X (gfx1250) — matrix operand order repaired and checked by reading, not compiled, not run on hardware
//
#include <hip/hip_runtime.h>
#include <hip/hip_bf16.h>
#include <math.h>

#define D      1024
#define BQ     256
#define NMEM   100000
#define TOPK   16
#define CHUNKS 250
#define CHROWS 400            // rows of memory per stage-1 workgroup
#define S1CAND (CHUNKS*TOPK)  // 4000 candidates per query row

typedef __attribute__((ext_vector_type(16))) __bf16       v16bf;
typedef __attribute__((ext_vector_type(8)))  float        v8f;
typedef __attribute__((ext_vector_type(8)))  unsigned int v8u;

union BF16Frag { v8u u; v16bf b; };

#if defined(__has_builtin)
#if __has_builtin(__builtin_amdgcn_cvt_pk_bf16_f32)
#define HAVE_HW_PK_BF16 1
#endif
#endif

// Pack two f32 into packed bf16 {hi, lo}. Preferred: hardware packed convert.
// Fallback: round-half-up via +0x8000 on the raw bits, then a single
// v_perm_b32 byte-select of the two high halves -> 3 VALU ops per pair.
__device__ __forceinline__ unsigned int pk_bf16(float lo, float hi) {
#ifdef HAVE_HW_PK_BF16
  auto t = __builtin_amdgcn_cvt_pk_bf16_f32(lo, hi);
  return __builtin_bit_cast(unsigned int, t);
#else
  unsigned int a = __builtin_bit_cast(unsigned int, lo) + 0x8000u;
  unsigned int b = __builtin_bit_cast(unsigned int, hi) + 0x8000u;
  // pool: sel 0-3 -> bytes of src1 (a), sel 4-7 -> bytes of src0 (b)
  // result = { b[31:24], b[23:16], a[31:24], a[23:16] } = {bf16(hi), bf16(lo)}
  return __builtin_amdgcn_perm(b, a, 0x07060302u);
#endif
}

__device__ __forceinline__ v8f zero8() {
  v8f z;
  for (int j = 0; j < 8; ++j) z[j] = 0.f;
  return z;
}

__device__ __forceinline__ v8f wmma_bf16(v16bf a, v16bf b, v8f c) {
  // (neg_a, A, neg_b, B, c_mod, C, reuse_a, reuse_b)
  return __builtin_amdgcn_wmma_f32_16x16x32_bf16(false, a, false, b, (short)0, c,
                                                 false, false);
}

// A fragment (16x32 bf16) built from f32 row-major source, converting in-register.
// Lanes 0-15: row=lane, K 0..15; lanes 16-31: row=lane-16, K 16..31.
__device__ __forceinline__ v16bf a_frag_f32(const float* __restrict__ A, int ld,
                                            int row0, int k0, int lane) {
  int m  = lane & 15;
  int kh = (lane & 16) ? 16 : 0;
  const float* p = A + (size_t)(row0 + m) * ld + k0 + kh;
  BF16Frag r;
#pragma unroll
  for (int i = 0; i < 4; ++i) r.u[i]     = pk_bf16(p[2*i],      p[2*i+1]);
#pragma unroll
  for (int i = 0; i < 4; ++i) r.u[4 + i] = pk_bf16(p[8 + 2*i], p[8 + 2*i + 1]);
  return r.b;
}

// A fragment from packed-bf16 (2 per uint) row-major source, ldu = D/2 uints.
__device__ __forceinline__ v16bf a_frag_bf(const unsigned int* __restrict__ A, int ldu,
                                           int row0, int k0, int lane) {
  int m  = lane & 15;
  int kh = (lane & 16) ? 8 : 0;
  const unsigned int* p = A + (size_t)(row0 + m) * ldu + (k0 >> 1) + kh;
  BF16Frag r;
#pragma unroll
  for (int i = 0; i < 4; ++i) r.u[i]     = p[i];
#pragma unroll
  for (int i = 0; i < 4; ++i) r.u[4 + i] = p[4 + i];
  return r.b;
}

// B fragment (32x16 bf16) for D = A*W^T style GEMMs: B(k,n) = W[n][k], W row-major,
// so each lane reads contiguous K from row n. Lanes 0-15: K 0..15, lanes 16-31: K 16..31.
__device__ __forceinline__ v16bf b_frag_f32(const float* __restrict__ W, int ld,
                                            int n0, int k0, int lane) {
  int n  = lane & 15;
  int kh = (lane & 16) ? 16 : 0;
  const float* p = W + (size_t)(n0 + n) * ld + k0 + kh;
  BF16Frag r;
#pragma unroll
  for (int i = 0; i < 8; ++i) r.u[i] = pk_bf16(p[2*i], p[2*i + 1]);
  return r.b;
}

// Same but from packed bf16 in LDS (ldu uints per row).
__device__ __forceinline__ v16bf b_frag_bf(const unsigned int* B, int ldu,
                                           int n0, int k0, int lane) {
  int n  = lane & 15;
  int kh = (lane & 16) ? 8 : 0;
  const unsigned int* p = B + (size_t)(n0 + n) * ldu + (k0 >> 1) + kh;
  BF16Frag r;
#pragma unroll
  for (int i = 0; i < 8; ++i) r.u[i] = p[i];
  return r.b;
}

// ---------------------------------------------------------------------------
// Kernel 1: Linear -> LayerNorm -> GELU(exact) -> Linear -> L2 normalize.
// One workgroup (8 waves) per 16-row tile of the batch; each wave owns 8 N-tiles.
// ---------------------------------------------------------------------------
__global__ __launch_bounds__(256) void es_mlp_kernel(
    const float* __restrict__ query, const float* __restrict__ W1,
    const float* __restrict__ b1, const float* __restrict__ ln_g,
    const float* __restrict__ ln_b, const float* __restrict__ W2,
    const float* __restrict__ b2, float* __restrict__ refined_out,
    unsigned int* __restrict__ refined_bf) {
  extern __shared__ char smem[];
  float*        hS  = (float*)smem;                       // 16 x 1024 f32 (64KB)
  unsigned int* hbS = (unsigned int*)(smem + 16 * D * 4); // 16 x 512 packed bf16 (32KB)

  const int tid  = threadIdx.x;
  const int wave = tid >> 5;
  const int lane = tid & 31;
  const int row0 = blockIdx.x * 16;

  // ---- GEMM1: h = query @ W1^T + b1 ----
  {
    v8f acc[8];
#pragma unroll
    for (int t = 0; t < 8; ++t) acc[t] = zero8();
    for (int k0 = 0; k0 < D; k0 += 32) {
      v16bf a = a_frag_f32(query, D, row0, k0, lane);
#pragma unroll
      for (int t = 0; t < 8; ++t) {
        v16bf b = b_frag_f32(W1, D, (wave * 8 + t) * 16, k0, lane);
        acc[t] = wmma_bf16(a, b, acc[t]);
      }
    }
    // C layout: VGPR j, lanes 0-15 -> (M=j, N=lane); lanes 16-31 -> (M=j+8, N=lane-16)
    int n = lane & 15, mh = (lane & 16) ? 8 : 0;
#pragma unroll
    for (int t = 0; t < 8; ++t) {
      int n0 = (wave * 8 + t) * 16;
      float bias = b1[n0 + n];
#pragma unroll
      for (int j = 0; j < 8; ++j) hS[(mh + j) * D + n0 + n] = acc[t][j] + bias;
    }
  }
  __syncthreads();

  // ---- LayerNorm + exact GELU (each wave handles 2 rows) ----
  for (int rr = 0; rr < 2; ++rr) {
    int r = wave * 2 + rr;
    float s = 0.f, ss = 0.f;
    for (int i = lane; i < D; i += 32) {
      float x = hS[r * D + i];
      s += x; ss += x * x;
    }
#pragma unroll
    for (int off = 16; off > 0; off >>= 1) {
      s  += __shfl_xor(s, off, 32);
      ss += __shfl_xor(ss, off, 32);
    }
    float mu   = s * (1.f / D);
    float var  = ss * (1.f / D) - mu * mu;
    float rstd = rsqrtf(var + 1e-5f);
    for (int i = lane; i < D; i += 32) {
      float x = (hS[r * D + i] - mu) * rstd * ln_g[i] + ln_b[i];
      hS[r * D + i] = 0.5f * x * (1.f + erff(x * 0.70710678118654752f));
    }
  }
  __syncthreads();
  for (int i = tid; i < 16 * (D / 2); i += 256)
    hbS[i] = pk_bf16(hS[2 * i], hS[2 * i + 1]);
  __syncthreads();

  // ---- GEMM2: y = gelu(h) @ W2^T + b2 ----
  {
    v8f acc[8];
#pragma unroll
    for (int t = 0; t < 8; ++t) acc[t] = zero8();
    for (int k0 = 0; k0 < D; k0 += 32) {
      v16bf a = a_frag_bf(hbS, D / 2, 0, k0, lane);
#pragma unroll
      for (int t = 0; t < 8; ++t) {
        v16bf b = b_frag_f32(W2, D, (wave * 8 + t) * 16, k0, lane);
        acc[t] = wmma_bf16(a, b, acc[t]);
      }
    }
    int n = lane & 15, mh = (lane & 16) ? 8 : 0;
#pragma unroll
    for (int t = 0; t < 8; ++t) {
      int n0 = (wave * 8 + t) * 16;
      float bias = b2[n0 + n];
#pragma unroll
      for (int j = 0; j < 8; ++j) hS[(mh + j) * D + n0 + n] = acc[t][j] + bias;
    }
  }
  __syncthreads();

  // ---- L2 normalize, emit f32 result + packed bf16 copy for the sims GEMM ----
  for (int rr = 0; rr < 2; ++rr) {
    int r = wave * 2 + rr;
    float ss = 0.f;
    for (int i = lane; i < D; i += 32) { float x = hS[r * D + i]; ss += x * x; }
#pragma unroll
    for (int off = 16; off > 0; off >>= 1) ss += __shfl_xor(ss, off, 32);
    float inv = 1.f / fmaxf(sqrtf(ss), 1e-12f);
    for (int i = lane; i < D; i += 32) {
      float y = hS[r * D + i] * inv;
      refined_out[(size_t)(row0 + r) * D + i] = y;
      hS[r * D + i] = y;
    }
  }
  __syncthreads();
  for (int i = tid; i < 16 * (D / 2); i += 256) {
    int r = i / (D / 2);
    refined_bf[(size_t)(row0 + r) * (D / 2) + (i % (D / 2))] =
        pk_bf16(hS[2 * i], hS[2 * i + 1]);
  }
}

// ---------------------------------------------------------------------------
// Kernel 2: sims = refined @ norm(memory)^T fused with per-chunk top-16.
// 16 waves; wave w owns query rows [16w,16w+16). WG streams 64 memory rows at a
// time into LDS (bf16), computes inv-norms on the fly, runs 4 N-tiles per
// A-fragment load, scales by inv-norm, and keeps a register-resident top-16.
// ---------------------------------------------------------------------------
__global__ __launch_bounds__(512) void es_sims_topk_kernel(
    const float* __restrict__ memory, const unsigned int* __restrict__ refined_bf,
    float* __restrict__ cand_val, int* __restrict__ cand_idx) {
  extern __shared__ char smem[];
  unsigned int* ldsB = (unsigned int*)smem;                 // 64 x 512 uints (128KB)
  float* invn = (float*)(smem + 64 * (D / 2) * 4);          // 64 f32
  float* psum = invn + 64;                                  // 64 x 8 partial sumsq
  float* xscr = psum + 64 * 8;                              // 16 waves x 256 f32

  const int tid  = threadIdx.x;
  const int wave = tid >> 5;
  const int lane = tid & 31;
  const int nbase = blockIdx.x * CHROWS;

  float kv[TOPK]; int ki[TOPK];
#pragma unroll
  for (int j = 0; j < TOPK; ++j) { kv[j] = -3.0e38f; ki[j] = 0; }
  float kmin = -3.0e38f; int kminpos = 0;

  for (int pass0 = 0; pass0 < CHROWS; pass0 += 64) {
    const int rem    = CHROWS - pass0;
    const int nrows  = rem < 64 ? rem : 64;
    const int ntiles = nrows >> 4;
    __syncthreads();
    // Load 64 rows, pack bf16 into LDS, deterministic per-row sumsq (8 partials/row).
    {
      int r = tid >> 3, seg = tid & 7;   // 8 threads per row x 64 rows
      if (r < nrows) {
        const float* src = memory + (size_t)(nbase + pass0 + r) * D + seg * 128;
        unsigned int* dst = ldsB + r * (D / 2) + seg * 64;
        float ss = 0.f;
#pragma unroll
        for (int i = 0; i < 64; ++i) {
          float x0 = src[2 * i], x1 = src[2 * i + 1];
          ss += x0 * x0 + x1 * x1;
          dst[i] = pk_bf16(x0, x1);
        }
        psum[r * 8 + seg] = ss;
      }
      if (pass0 + 64 < CHROWS && r < 64)  // prefetch next pass (global_prefetch_b8)
        __builtin_prefetch(memory + (size_t)(nbase + pass0 + 64 + r) * D + seg * 128, 0, 1);
    }
    __syncthreads();
    if (tid < 64) {
      float s = 0.f;
#pragma unroll
      for (int j = 0; j < 8; ++j) s += psum[tid * 8 + j];
      invn[tid] = 1.f / fmaxf(sqrtf(s), 1e-12f);
    }
    __syncthreads();

    // WMMA: 4 N-tiles per A-fragment load
    v8f acc[4];
#pragma unroll
    for (int t = 0; t < 4; ++t) acc[t] = zero8();
    for (int k0 = 0; k0 < D; k0 += 32) {
      v16bf a = a_frag_bf(refined_bf, D / 2, wave * 16, k0, lane);
#pragma unroll
      for (int t = 0; t < 4; ++t)
        if (t < ntiles) {
          v16bf b = b_frag_bf(ldsB, D / 2, t * 16, k0, lane);
          acc[t] = wmma_bf16(a, b, acc[t]);
        }
    }

    // Scale by inv-norm, transpose through per-wave LDS scratch, merge top-16.
    int n = lane & 15, mh = (lane & 16) ? 8 : 0;
    for (int t = 0; t < 4; ++t) {
      if (t < ntiles) {
        float sc = invn[t * 16 + n];
        float* scr = xscr + wave * 256;
#pragma unroll
        for (int j = 0; j < 8; ++j) scr[(mh + j) * 16 + n] = acc[t][j] * sc;
      }
      __syncthreads();
      if (t < ntiles && lane < 16) {
        const float* scr = xscr + wave * 256;
        int colbase = nbase + pass0 + t * 16;
#pragma unroll
        for (int c = 0; c < 16; ++c) {
          float v = scr[lane * 16 + c];
          if (v > kmin) {
            int idx = colbase + c;
#pragma unroll
            for (int j = 0; j < TOPK; ++j)
              if (j == kminpos) { kv[j] = v; ki[j] = idx; }
            kmin = kv[0]; kminpos = 0;
#pragma unroll
            for (int j = 1; j < TOPK; ++j)
              if (kv[j] < kmin) { kmin = kv[j]; kminpos = j; }
          }
        }
      }
      __syncthreads();
    }
  }

  if (lane < 16) {
    int row = wave * 16 + lane;
    float* vo = cand_val + (size_t)row * S1CAND + blockIdx.x * TOPK;
    int*   io = cand_idx + (size_t)row * S1CAND + blockIdx.x * TOPK;
#pragma unroll
    for (int j = 0; j < TOPK; ++j) { vo[j] = kv[j]; io[j] = ki[j]; }
  }
}

// ---------------------------------------------------------------------------
// Kernel 3: per-row merge of 4000 candidates -> sorted top-16, then gather the
// 16 memory rows, L2-normalize, and write `retrieved`.
// ---------------------------------------------------------------------------
__global__ __launch_bounds__(256) void es_merge_gather_kernel(
    const float* __restrict__ cand_val, const int* __restrict__ cand_idx,
    const float* __restrict__ memory, float* __restrict__ topk_vals,
    int* __restrict__ topk_idx, float* __restrict__ retrieved) {
  extern __shared__ char smem[];
  float* cv   = (float*)smem;            // 4000
  int*   ci   = (int*)(cv + S1CAND);     // 4000
  float* rv   = (float*)(ci + S1CAND);   // 256
  int*   rp   = (int*)(rv + 256);        // 256
  float* winv = (float*)(rp + 256);      // 16
  int*   wini = (int*)(winv + 16);       // 16

  const int tid = threadIdx.x, row = blockIdx.x;
  for (int i = tid; i < S1CAND; i += 256) {
    cv[i] = cand_val[(size_t)row * S1CAND + i];
    ci[i] = cand_idx[(size_t)row * S1CAND + i];
  }
  __syncthreads();

  for (int it = 0; it < TOPK; ++it) {
    float bv = -3.0e38f; int bp = 0;
    for (int i = tid; i < S1CAND; i += 256) {
      float v = cv[i];
      if (v > bv) { bv = v; bp = i; }
    }
    rv[tid] = bv; rp[tid] = bp;
    __syncthreads();
    for (int off = 128; off > 0; off >>= 1) {
      if (tid < off && rv[tid + off] > rv[tid]) {
        rv[tid] = rv[tid + off]; rp[tid] = rp[tid + off];
      }
      __syncthreads();
    }
    if (tid == 0) {
      int p = rp[0];
      winv[it] = rv[0];
      wini[it] = ci[p];
      cv[p] = -3.4e38f;  // knock out winner
    }
    __syncthreads();
  }

  if (tid < TOPK) {
    topk_vals[(size_t)row * TOPK + tid] = winv[tid];
    topk_idx [(size_t)row * TOPK + tid] = wini[tid];
  }
  __syncthreads();

  for (int j = 0; j < TOPK; ++j) {
    const float* src = memory + (size_t)wini[j] * D;
    float ss = 0.f;
    for (int i = tid; i < D; i += 256) { float x = src[i]; ss += x * x; }
    rv[tid] = ss;
    __syncthreads();
    for (int off = 128; off > 0; off >>= 1) {
      if (tid < off) rv[tid] += rv[tid + off];
      __syncthreads();
    }
    float inv = 1.f / fmaxf(sqrtf(rv[0]), 1e-12f);
    __syncthreads();
    for (int i = tid; i < D; i += 256)
      retrieved[((size_t)row * TOPK + j) * D + i] = src[i] * inv;
  }
}

extern "C" void kernel_launch(void* const* d_in, const int* in_sizes, int n_in,
                              void* d_out, int out_size, void* d_ws, size_t ws_size,
                              hipStream_t stream) {
  (void)in_sizes; (void)n_in; (void)out_size; (void)ws_size;
  const float* query  = (const float*)d_in[0];
  const float* W1     = (const float*)d_in[1];
  const float* b1     = (const float*)d_in[2];
  const float* ln_g   = (const float*)d_in[3];
  const float* ln_b   = (const float*)d_in[4];
  const float* W2     = (const float*)d_in[5];
  const float* b2     = (const float*)d_in[6];
  const float* memory = (const float*)d_in[7];
  // d_in[8] = k (== 16, compile-time TOPK)

  float* out       = (float*)d_out;
  float* refined   = out;                              // 256*1024
  float* tk_vals   = out + (size_t)BQ * D;             // 256*16
  int*   tk_idx    = (int*)(out + (size_t)BQ * D + BQ * TOPK);
  float* retrieved = out + (size_t)BQ * D + 2 * BQ * TOPK;  // 256*16*1024

  char* ws = (char*)d_ws;
  unsigned int* refined_bf = (unsigned int*)ws;                       // 512KB
  float* cand_val = (float*)(ws + (size_t)BQ * (D / 2) * 4);          // 4MB
  int*   cand_idx = (int*)(ws + (size_t)BQ * (D / 2) * 4 +
                           (size_t)BQ * S1CAND * 4);                  // 4MB

  size_t lds1 = (size_t)16 * D * 4 + (size_t)16 * (D / 2) * 4;  // 96KB
  es_mlp_kernel<<<dim3(BQ / 16), dim3(256), lds1, stream>>>(
      query, W1, b1, ln_g, ln_b, W2, b2, refined, refined_bf);

  size_t lds2 = (size_t)64 * (D / 2) * 4 + 64 * 4 + 64 * 8 * 4 +
                (size_t)16 * 256 * 4;  // ~146KB (of 320KB/WGP)
  es_sims_topk_kernel<<<dim3(CHUNKS), dim3(512), lds2, stream>>>(
      memory, refined_bf, cand_val, cand_idx);

  size_t lds3 = (size_t)S1CAND * 8 + 256 * 8 + TOPK * 8;
  es_merge_gather_kernel<<<dim3(BQ), dim3(256), lds3, stream>>>(
      cand_val, cand_idx, memory, tk_vals, tk_idx, retrieved);
}
